// VectorQuantizer_48962627174912
// MI455X (gfx1250) — matrix operand order repaired and checked by reading, not compiled, not run on hardware
//
#include <hip/hip_runtime.h>
#include <hip/hip_bf16.h>

typedef unsigned short u16;
typedef __bf16 bf16_t;
typedef bf16_t bf16x16 __attribute__((ext_vector_type(16)));
typedef float f32x8 __attribute__((ext_vector_type(8)));
typedef int v4i __attribute__((ext_vector_type(4)));

#define NB 32
#define DD 256
#define HW 1024
#define NN 32768      // NB*HW rows
#define KK 4096       // codebook entries
#define LDS_STRIDE 264  // 256 + 8 u16 pad -> conflict-free ds_load_b128

union Frag16 { bf16x16 v; uint4 q[2]; };

// ---- bf16 split helpers (RNE) ----
__device__ __forceinline__ u16 f32_to_bf16_rne(float x) {
    unsigned u = __float_as_uint(x);
    return (u16)((u + 0x7FFFu + ((u >> 16) & 1u)) >> 16);
}
__device__ __forceinline__ float bf16_bits_to_f32(u16 h) {
    return __uint_as_float(((unsigned)h) << 16);
}

// ---- async global->LDS copy (CDNA5 path w/ fallback) ----
#if __has_builtin(__builtin_amdgcn_global_load_async_to_lds_b128)
#define HAVE_ASYNC_LDS 1
#endif

__device__ __forceinline__ void cp16(void* lds_dst, const void* g_src) {
#ifdef HAVE_ASYNC_LDS
    typedef __attribute__((address_space(1))) v4i GV4;
    typedef __attribute__((address_space(3))) v4i LV4;
    __builtin_amdgcn_global_load_async_to_lds_b128((GV4*)g_src, (LV4*)lds_dst, 0, 0);
#else
    *(uint4*)lds_dst = *(const uint4*)g_src;
#endif
}
__device__ __forceinline__ void cp16_wait() {
#ifdef HAVE_ASYNC_LDS
    asm volatile("s_wait_asynccnt 0" ::: "memory");
#endif
}

// ---- Prep: transpose z (B,D,H,W)->(N,D) and split to bf16 hi/lo ----
__global__ void pack_z(const float* __restrict__ z, u16* __restrict__ zhi, u16* __restrict__ zlo) {
    int n = blockIdx.x, d = threadIdx.x;
    int b = n >> 10, hw = n & 1023;
    float x = z[(((size_t)(b * DD + d)) << 10) + hw];
    u16 hi = f32_to_bf16_rne(x);
    float lo = x - bf16_bits_to_f32(hi);
    u16 lo16 = f32_to_bf16_rne(lo);
    size_t o = (size_t)n * DD + d;
    zhi[o] = hi;
    zlo[o] = lo16;
}

// ---- Prep: split embedding + ||e||^2 in fp32 ----
__global__ void pack_e(const float* __restrict__ E, u16* __restrict__ ehi,
                       u16* __restrict__ elo, float* __restrict__ enorm) {
    int k = blockIdx.x, d = threadIdx.x;
    float x = E[(size_t)k * DD + d];
    u16 hi = f32_to_bf16_rne(x);
    float lo = x - bf16_bits_to_f32(hi);
    size_t o = (size_t)k * DD + d;
    ehi[o] = hi;
    elo[o] = f32_to_bf16_rne(lo);
    __shared__ float red[256];
    red[d] = x * x;
    __syncthreads();
    for (int s = 128; s > 0; s >>= 1) {
        if (d < s) red[d] += red[d + s];
        __syncthreads();
    }
    if (d == 0) enorm[k] = red[0];
}

// ---- Fused split-bf16 WMMA GEMM + argmin ----
// Block: 256 threads = 8 waves; wave owns 16 rows; block covers 128 rows.
// K loop in 64-col chunks, chunk staged in LDS (async), 4 N-tiles per wave.
__global__ __launch_bounds__(256) void vq_argmin(
    const u16* __restrict__ zhi, const u16* __restrict__ zlo,
    const u16* __restrict__ ehi, const u16* __restrict__ elo,
    const float* __restrict__ enorm, float* __restrict__ idx_out) {
    __shared__ u16 lds_hi[64 * LDS_STRIDE];
    __shared__ u16 lds_lo[64 * LDS_STRIDE];

    const int tid = threadIdx.x;
    const int wave = tid >> 5;
    const int lane = tid & 31;
    const int lane15 = lane & 15;
    const int halfid = lane >> 4;
    const int rowBase = blockIdx.x * 128 + wave * 16;

    const u16* arow_hi = zhi + (size_t)(rowBase + lane15) * DD;
    const u16* arow_lo = zlo + (size_t)(rowBase + lane15) * DD;

    float minv[8];
    int mini[8];
#pragma unroll
    for (int r = 0; r < 8; ++r) { minv[r] = 3.4e38f; mini[r] = 0; }

    for (int colBase = 0; colBase < KK; colBase += 64) {
        __syncthreads();  // previous chunk's LDS reads done
        {
            const uint4* ghi = (const uint4*)(ehi + (size_t)colBase * DD);
            const uint4* glo = (const uint4*)(elo + (size_t)colBase * DD);
#pragma unroll
            for (int i = 0; i < 8; ++i) {
                int j = tid + i * 256;      // 0..2047 ; row = j/32 (32 uint4 per 256-d row)
                int row = j >> 5, c = j & 31;
                cp16((uint4*)(lds_hi + (size_t)row * LDS_STRIDE) + c, ghi + j);
                cp16((uint4*)(lds_lo + (size_t)row * LDS_STRIDE) + c, glo + j);
            }
            cp16_wait();
        }
        __syncthreads();

        f32x8 acc0 = {}, acc1 = {}, acc2 = {}, acc3 = {};
#pragma unroll
        for (int kb = 0; kb < DD; kb += 32) {
            // A fragment (16x32 bf16): lane row = lane%16; K = halfid*16 + j (contiguous 32B)
            Frag16 ah, al;
            const uint4* pah = (const uint4*)(arow_hi + kb + halfid * 16);
            const uint4* pal = (const uint4*)(arow_lo + kb + halfid * 16);
            ah.q[0] = pah[0]; ah.q[1] = pah[1];
            al.q[0] = pal[0]; al.q[1] = pal[1];
#pragma unroll
            for (int t = 0; t < 4; ++t) {
                // B fragment (32x16 bf16): N = lane%16, K = halfid*16 + j (contiguous 32B)
                Frag16 bh, bl;
                const uint4* pbh = (const uint4*)(lds_hi + (size_t)(t * 16 + lane15) * LDS_STRIDE + kb + halfid * 16);
                const uint4* pbl = (const uint4*)(lds_lo + (size_t)(t * 16 + lane15) * LDS_STRIDE + kb + halfid * 16);
                bh.q[0] = pbh[0]; bh.q[1] = pbh[1];
                bl.q[0] = pbl[0]; bl.q[1] = pbl[1];
                f32x8& acc = (t == 0) ? acc0 : (t == 1) ? acc1 : (t == 2) ? acc2 : acc3;
                acc = __builtin_amdgcn_wmma_f32_16x16x32_bf16(false, ah.v, false, bh.v, (short)0, acc, false, false);
                acc = __builtin_amdgcn_wmma_f32_16x16x32_bf16(false, ah.v, false, bl.v, (short)0, acc, false, false);
                acc = __builtin_amdgcn_wmma_f32_16x16x32_bf16(false, al.v, false, bh.v, (short)0, acc, false, false);
            }
        }
        // score = ||e||^2 - 2*(z.e)   (||z||^2 constant per row, dropped)
#pragma unroll
        for (int t = 0; t < 4; ++t) {
            const f32x8& acc = (t == 0) ? acc0 : (t == 1) ? acc1 : (t == 2) ? acc2 : acc3;
            int col = colBase + t * 16 + lane15;
            float en = enorm[col];
#pragma unroll
            for (int r = 0; r < 8; ++r) {
                float dist = en - 2.0f * acc[r];
                if (dist < minv[r]) { minv[r] = dist; mini[r] = col; }
            }
        }
    }

    // cross-lane argmin within each 16-lane half (rows disjoint between halves)
#pragma unroll
    for (int off = 1; off < 16; off <<= 1) {
#pragma unroll
        for (int r = 0; r < 8; ++r) {
            float ov = __shfl_xor(minv[r], off, 32);
            int oi = __shfl_xor(mini[r], off, 32);
            if (ov < minv[r] || (ov == minv[r] && oi < mini[r])) { minv[r] = ov; mini[r] = oi; }
        }
    }
    if (lane15 == 0) {
#pragma unroll
        for (int r = 0; r < 8; ++r)
            idx_out[rowBase + halfid * 8 + r] = (float)mini[r];  // row = r + 8*(lane/16)
    }
}

// ---- Gather codebook rows, scatter to (B,D,H,W), per-row loss partial ----
__global__ void gather_loss(const float* __restrict__ z, const float* __restrict__ E,
                            const float* __restrict__ idxf, float* __restrict__ zq_out,
                            float* __restrict__ rowsum) {
    int n = blockIdx.x, d = threadIdx.x;
    int idx = (int)idxf[n];
    float e = E[(size_t)idx * DD + d];
    int b = n >> 10, hw = n & 1023;
    size_t off = (((size_t)(b * DD + d)) << 10) + hw;
    float zv = z[off];
    zq_out[off] = e;  // straight-through forward value == zq
    float diff = e - zv;
    __shared__ float red[256];
    red[d] = diff * diff;
    __syncthreads();
    for (int s = 128; s > 0; s >>= 1) {
        if (d < s) red[d] += red[d + s];
        __syncthreads();
    }
    if (d == 0) rowsum[n] = red[0];
}

// ---- Final scalar: vq_loss = (1 + 0.25) * mean((zq - ze)^2) ----
__global__ void final_loss(const float* __restrict__ rowsum, float* __restrict__ out_loss) {
    __shared__ float red[256];
    float s = 0.0f;
    for (int i = threadIdx.x; i < NN; i += 256) s += rowsum[i];
    red[threadIdx.x] = s;
    __syncthreads();
    for (int t = 128; t > 0; t >>= 1) {
        if (threadIdx.x < t) red[threadIdx.x] += red[threadIdx.x + t];
        __syncthreads();
    }
    if (threadIdx.x == 0) *out_loss = 1.25f * red[0] / 8388608.0f;
}

extern "C" void kernel_launch(void* const* d_in, const int* in_sizes, int n_in,
                              void* d_out, int out_size, void* d_ws, size_t ws_size,
                              hipStream_t stream) {
    const float* z = (const float*)d_in[0];       // (32,256,32,32)
    const float* emb = (const float*)d_in[1];     // (4096,256)
    float* out = (float*)d_out;
    float* zq_out = out;                          // 8388608 floats
    float* idx_out = out + 8388608;               // 32768 floats (indices as float)
    float* loss_out = out + 8388608 + 32768;      // 1 float

    char* ws = (char*)d_ws;
    u16* zf_hi = (u16*)ws;                                      // 16 MB
    u16* zf_lo = (u16*)(ws + 16777216);                         // 16 MB
    u16* e_hi = (u16*)(ws + 2 * 16777216);                      // 2 MB
    u16* e_lo = (u16*)(ws + 2 * 16777216 + 2097152);            // 2 MB
    float* e_norm = (float*)(ws + 2 * 16777216 + 2 * 2097152);  // 16 KB
    float* rowsum = (float*)(ws + 2 * 16777216 + 2 * 2097152 + 16384);  // 128 KB

    pack_z<<<NN, 256, 0, stream>>>(z, zf_hi, zf_lo);
    pack_e<<<KK, 256, 0, stream>>>(emb, e_hi, e_lo, e_norm);
    vq_argmin<<<NN / 128, 256, 0, stream>>>(zf_hi, zf_lo, e_hi, e_lo, e_norm, idx_out);
    gather_loss<<<NN, 256, 0, stream>>>(z, emb, idx_out, zq_out, rowsum);
    final_loss<<<1, 256, 0, stream>>>(rowsum, loss_out);
}
